// TransformerDecoder_30073361007032
// MI455X (gfx1250) — matrix operand order repaired and checked by reading, not compiled, not run on hardware
//
#include <hip/hip_runtime.h>
#include <math.h>

// ---------------------------------------------------------------------------
// Problem constants (from reference)
// ---------------------------------------------------------------------------
static constexpr int B_  = 4;
static constexpr int S_  = 2048;
static constexpr int V_  = 50257;
static constexpr int D_  = 1024;
static constexpr int DK_ = 1024;
static constexpr int DV_ = 1024;
static constexpr int DH_ = 4096;

// ---------------------------------------------------------------------------
// bf16 plumbing
// ---------------------------------------------------------------------------
typedef __bf16 bf16_t;
typedef __attribute__((ext_vector_type(16))) bf16_t v16bf;
typedef __attribute__((ext_vector_type(8)))  float  v8f;
typedef __attribute__((ext_vector_type(4)))  unsigned int v4u;
typedef __attribute__((ext_vector_type(4)))  int v4i;
typedef __attribute__((ext_vector_type(8)))  int v8i;

struct Q2 { uint4 lo; uint4 hi; };   // 32 bytes == v16bf

__device__ __forceinline__ bf16_t f2bf(float f) {
    unsigned u = __float_as_uint(f);
    unsigned r = (u + 0x7FFFu + ((u >> 16) & 1u)) >> 16;
    unsigned short s = (unsigned short)r;
    return __builtin_bit_cast(bf16_t, s);
}

#if defined(__has_builtin)
#if __has_builtin(__builtin_amdgcn_tensor_load_to_lds)
#define HAS_TDM 1
#endif
#endif

__device__ __forceinline__ void wait_tensorcnt0() {
#if defined(__has_builtin)
#if __has_builtin(__builtin_amdgcn_s_wait_tensorcnt)
    __builtin_amdgcn_s_wait_tensorcnt(0);
    return;
#endif
#endif
    asm volatile("s_wait_tensorcnt 0x0" ::: "memory");
}

#ifdef HAS_TDM
// Issue a TDM load of a 2-D tile (tile_d0 x tile_d1, 2-byte elements) from a
// row-major tensor (inner length tensor_d0, stride0 elements per line) into
// LDS at lds_off.  TDM inserts 16B of LDS padding per 64B stored
// (pad_interval=3 -> 64B, pad_amount=3 -> 4 DWORDs): LDS row stride becomes
// 80B = 40 elements, which makes the WMMA fragment reads bank-conflict-free.
// D# layout per CDNA5 ISA 8.3/8.4.
__device__ __forceinline__ void tdm_load_tile_2d(unsigned lds_off,
                                                 const void* gptr,
                                                 unsigned tensor_d0,
                                                 unsigned tensor_d1,
                                                 unsigned tile_d0,
                                                 unsigned tile_d1,
                                                 unsigned stride0) {
    unsigned long long ga = (unsigned long long)(uintptr_t)gptr;
    v4u g0;
    g0[0] = 1u;                                        // count=1 (valid user D#)
    g0[1] = lds_off;                                   // lds_addr (bytes)
    g0[2] = (unsigned)(ga & 0xffffffffu);              // global_addr[31:0]
    g0[3] = (unsigned)((ga >> 32) & 0x01ffffffu)       // global_addr[56:32]
          | 0x80000000u;                               // type=2 ("image")
    v8i g1;
    g1[0] = (int)((1u << 16)                           // data_size = 2 bytes
                | (1u << 20)                           // pad_enable
                | (3u << 22)                           // pad_interval: 16 DW (64B)
                | (3u << 25));                         // pad_amount: 4 DW (16B)
    g1[1] = (int)((tensor_d0 & 0xffffu) << 16);        // tensor_dim0[15:0]
    g1[2] = (int)((tensor_d0 >> 16) | ((tensor_d1 & 0xffffu) << 16));
    g1[3] = (int)((tensor_d1 >> 16) | (tile_d0 << 16));
    g1[4] = (int)(tile_d1 & 0xffffu);                  // tile_dim1 (tile_dim2=0)
    g1[5] = (int)stride0;                              // tensor_dim0_stride[31:0]
    g1[6] = 0;
    g1[7] = 0;
    v4i z4 = {0, 0, 0, 0};
#if __clang_major__ >= 23
    v8i z8 = {0, 0, 0, 0, 0, 0, 0, 0};
    __builtin_amdgcn_tensor_load_to_lds(g0, g1, z4, z4, z8, 0);
#else
    __builtin_amdgcn_tensor_load_to_lds(g0, g1, z4, z4, 0);
#endif
}
#endif  // HAS_TDM

// ---------------------------------------------------------------------------
// K1b: f32 [K,N] -> bf16 transposed [N,K]  (tiled LDS transpose, coalesced)
// ---------------------------------------------------------------------------
__global__ __launch_bounds__(256)
void cvt_transpose_kernel(const float* __restrict__ src,
                          bf16_t* __restrict__ dst, int K, int N) {
    __shared__ float tile[32][33];
    int k0 = blockIdx.y * 32, n0 = blockIdx.x * 32;
    int tx = threadIdx.x & 31, ty = threadIdx.x >> 5;   // 32 x 8
#pragma unroll
    for (int i = 0; i < 4; ++i) {
        int kr = ty + i * 8;
        tile[kr][tx] = src[(size_t)(k0 + kr) * N + n0 + tx];
    }
    __syncthreads();
#pragma unroll
    for (int i = 0; i < 4; ++i) {
        int nr = ty + i * 8;
        dst[(size_t)(n0 + nr) * K + k0 + tx] = f2bf(tile[tx][nr]);
    }
}

// ---------------------------------------------------------------------------
// K2: h = emb[x] + pos   (writes f32 and bf16 copies)
// ---------------------------------------------------------------------------
__global__ void embed_kernel(const int* __restrict__ x,
                             const float* __restrict__ emb,
                             const float* __restrict__ pos,
                             float* __restrict__ hf,
                             bf16_t* __restrict__ hb,
                             int S, int D) {
    int row = blockIdx.x;            // b*S + s
    int s   = row % S;
    long long tok = (long long)x[row];
    const float* er = emb + (size_t)tok * D;
    const float* pr = pos + (size_t)s   * D;
    for (int d = threadIdx.x; d < D; d += blockDim.x) {
        float v = er[d] + pr[d];
        hf[(size_t)row * D + d] = v;
        hb[(size_t)row * D + d] = f2bf(v);
    }
}

// ---------------------------------------------------------------------------
// K3: tiled bf16 WMMA GEMM:  C[M,N] = A[M,K] * Bt[N,K]^T + bias[N]
//     Both operands row-major with contiguous K, staged by the Tensor Data
//     Mover into double-buffered LDS: wave 0 issues the DMA for tile t+1,
//     all waves compute tile t (8 x v_wmma_f32_16x16x32_bf16), then wave 0
//     drains TENSORcnt and the block barriers -> DMA overlaps the math.
//     Block tile 128x128x32, 8 waves, wave tile 32x64.
// ---------------------------------------------------------------------------
#define BM 128
#define BN 128
#define BKK 32
#define LDS_STRIDE 40     // 32 elems + 8 pad (16B per 64B row)
#define TILE_ELEMS (BM * LDS_STRIDE)

__device__ __forceinline__ void wmma_tile_step(const bf16_t* __restrict__ As,
                                               const bf16_t* __restrict__ Bs,
                                               v8f acc[2][4],
                                               int wm0, int wn0,
                                               int half, int lr) {
    // A lane needs K = half*8 + [0..7] and 16 + half*8 + [0..7]
    // B lane needs K = half*8 + [0..7] and 16 + half*8 + [0..7] (ISA 7.12.2 layouts)
    v16bf af[2];
#pragma unroll
    for (int mi = 0; mi < 2; ++mi) {
        int row = wm0 + mi * 16 + lr;
        Q2 q;
        q.lo = *(const uint4*)&As[row * LDS_STRIDE + half * 8];
        q.hi = *(const uint4*)&As[row * LDS_STRIDE + 16 + half * 8];
        af[mi] = __builtin_bit_cast(v16bf, q);
    }
    v16bf bfv[4];
#pragma unroll
    for (int ni = 0; ni < 4; ++ni) {
        int col = wn0 + ni * 16 + lr;
        Q2 q;
        q.lo = *(const uint4*)&Bs[col * LDS_STRIDE + half * 8];
        q.hi = *(const uint4*)&Bs[col * LDS_STRIDE + 16 + half * 8];
        bfv[ni] = __builtin_bit_cast(v16bf, q);
    }
#pragma unroll
    for (int mi = 0; mi < 2; ++mi)
#pragma unroll
        for (int ni = 0; ni < 4; ++ni)
            acc[mi][ni] = __builtin_amdgcn_wmma_f32_16x16x32_bf16(
                false, af[mi], false, bfv[ni],
                (short)0, acc[mi][ni], false, false);
}

__global__ __launch_bounds__(256)
void gemm_bf16_wmma(const bf16_t* __restrict__ A,
                    const bf16_t* __restrict__ Bt,
                    const float*  __restrict__ bias,
                    float* __restrict__ C,
                    int M, int N, int K) {
    __shared__ __align__(16) bf16_t As[2][TILE_ELEMS];
    __shared__ __align__(16) bf16_t Bs[2][TILE_ELEMS];

    const int tid  = threadIdx.x;
    const int wave = tid >> 5;
    const int lane = tid & 31;
    const int half = lane >> 4;
    const int lr   = lane & 15;

    const int m0  = blockIdx.y * BM;
    const int n0  = blockIdx.x * BN;
    const int wm0 = (wave >> 1) * 32;   // 4 waves along M
    const int wn0 = (wave & 1)  * 64;   // 2 waves along N

    v8f acc[2][4];
#pragma unroll
    for (int mi = 0; mi < 2; ++mi)
#pragma unroll
        for (int ni = 0; ni < 4; ++ni)
#pragma unroll
            for (int r = 0; r < 8; ++r) acc[mi][ni][r] = 0.0f;

    const int ntiles = K / BKK;

#ifdef HAS_TDM
    // prologue: DMA tile 0 into buffer 0
    if (wave == 0) {
        tdm_load_tile_2d((unsigned)(uintptr_t)(void*)As[0],
                         A + (size_t)m0 * K,
                         (unsigned)K, (unsigned)M, BKK, BM, (unsigned)K);
        tdm_load_tile_2d((unsigned)(uintptr_t)(void*)Bs[0],
                         Bt + (size_t)n0 * K,
                         (unsigned)K, (unsigned)N, BKK, BN, (unsigned)K);
        wait_tensorcnt0();
    }
    __syncthreads();

    for (int t = 0; t < ntiles; ++t) {
        const int cur = t & 1, nxt = cur ^ 1;
        // kick off DMA for tile t+1 while tile t is consumed
        if (wave == 0 && (t + 1) < ntiles) {
            size_t ko = (size_t)(t + 1) * BKK;
            tdm_load_tile_2d((unsigned)(uintptr_t)(void*)As[nxt],
                             A + (size_t)m0 * K + ko,
                             (unsigned)K, (unsigned)M, BKK, BM, (unsigned)K);
            tdm_load_tile_2d((unsigned)(uintptr_t)(void*)Bs[nxt],
                             Bt + (size_t)n0 * K + ko,
                             (unsigned)K, (unsigned)N, BKK, BN, (unsigned)K);
        }
        wmma_tile_step(As[cur], Bs[cur], acc, wm0, wn0, half, lr);
        if (wave == 0) wait_tensorcnt0();   // drain before anyone reads nxt
        __syncthreads();
    }
#else
    // fallback: vectorized global->LDS staging, single buffer
    for (int t = 0; t < ntiles; ++t) {
        int kt = t * BKK;
#pragma unroll
        for (int it = 0; it < 2; ++it) {
            int idx = tid + it * 256;          // 0..511
            int r   = idx >> 2;
            int c   = (idx & 3) * 8;
            uint4 av = {0u, 0u, 0u, 0u};
            if (m0 + r < M)
                av = *(const uint4*)(A + (size_t)(m0 + r) * K + kt + c);
            *(uint4*)&As[0][r * LDS_STRIDE + c] = av;
            uint4 bv = *(const uint4*)(Bt + (size_t)(n0 + r) * K + kt + c);
            *(uint4*)&Bs[0][r * LDS_STRIDE + c] = bv;
        }
        __syncthreads();
        wmma_tile_step(As[0], Bs[0], acc, wm0, wn0, half, lr);
        __syncthreads();
    }
#endif

    // epilogue: f32 C/D layout (lane lr = col, vgpr r + half*8 = row)
#pragma unroll
    for (int mi = 0; mi < 2; ++mi) {
#pragma unroll
        for (int ni = 0; ni < 4; ++ni) {
            int col = n0 + wn0 + ni * 16 + lr;
            float bsv = bias ? bias[col] : 0.0f;
#pragma unroll
            for (int r = 0; r < 8; ++r) {
                int row = m0 + wm0 + mi * 16 + half * 8 + r;
                if (row < M)
                    C[(size_t)row * N + col] = acc[mi][ni][r] + bsv;
            }
        }
    }
}

// ---------------------------------------------------------------------------
// K4: per-batch-row GEMV: out[b,n] = act( sum_k xrow[b,k]*W[k,N:n] + bias[n] )
//     W is streamed exactly once per launch -> non-temporal loads so the
//     205MB wout stream does not evict the reused k/v set from the 192MB L2.
// ---------------------------------------------------------------------------
__global__ void gemv_row_kernel(const float* __restrict__ xrows, long long ldx,
                                const float* __restrict__ W,
                                const float* __restrict__ bias,
                                float* __restrict__ out, int ldo,
                                int K, int N, int act) {
    extern __shared__ float xs[];
    int b = blockIdx.y;
    const float* xr = xrows + (size_t)b * ldx;
    for (int k = threadIdx.x; k < K; k += blockDim.x) xs[k] = xr[k];
    __syncthreads();

    int n = blockIdx.x * blockDim.x + threadIdx.x;
    if (n >= N) return;
    float acc = bias[n];
#pragma unroll 4
    for (int k = 0; k < K; ++k)
        acc = fmaf(xs[k], __builtin_nontemporal_load(&W[(size_t)k * N + n]), acc);
    if (act == 1)
        acc = 0.5f * acc * (1.0f + erff(acc * 0.70710678118654752f));
    out[(size_t)b * ldo + n] = acc;
}

// ---------------------------------------------------------------------------
// K5: scores[b,j] = (q[b,:] . k[b,j,:]) * scale   — one block per (j,b)
// ---------------------------------------------------------------------------
__global__ void scores_kernel(const float* __restrict__ q,
                              const float* __restrict__ k,
                              float* __restrict__ s,
                              int S, int DK, float scale) {
    int b = blockIdx.y, j = blockIdx.x, tid = threadIdx.x;
    const float* qr = q + (size_t)b * DK;
    const float* kr = k + ((size_t)b * S + j) * DK;
    float part = 0.0f;
    for (int d = tid; d < DK; d += 256) part = fmaf(qr[d], kr[d], part);
    __shared__ float red[256];
    red[tid] = part; __syncthreads();
    for (int off = 128; off > 0; off >>= 1) {
        if (tid < off) red[tid] += red[tid + off];
        __syncthreads();
    }
    if (tid == 0) s[(size_t)b * S + j] = red[0] * scale;
}

// ---------------------------------------------------------------------------
// K6: softmax over a full row of S (last-token row: causal mask all-true)
// ---------------------------------------------------------------------------
__global__ void softmax_kernel(const float* __restrict__ s,
                               float* __restrict__ a, int S) {
    int b = blockIdx.x, tid = threadIdx.x;
    __shared__ float red[256];
    __shared__ float bval;
    float mx = -3.402823466e38f;
    for (int j = tid; j < S; j += 256) mx = fmaxf(mx, s[(size_t)b * S + j]);
    red[tid] = mx; __syncthreads();
    for (int off = 128; off > 0; off >>= 1) {
        if (tid < off) red[tid] = fmaxf(red[tid], red[tid + off]);
        __syncthreads();
    }
    if (tid == 0) bval = red[0];
    __syncthreads();
    float m = bval, sum = 0.0f;
    for (int j = tid; j < S; j += 256) {
        float e = expf(s[(size_t)b * S + j] - m);
        a[(size_t)b * S + j] = e;
        sum += e;
    }
    __syncthreads();
    red[tid] = sum; __syncthreads();
    for (int off = 128; off > 0; off >>= 1) {
        if (tid < off) red[tid] += red[tid + off];
        __syncthreads();
    }
    if (tid == 0) bval = 1.0f / red[0];
    __syncthreads();
    float inv = bval;
    for (int j = tid; j < S; j += 256) a[(size_t)b * S + j] *= inv;
}

// ---------------------------------------------------------------------------
// K7: sa[b,d] = sum_j attn[b,j] * v[b,j,d]   — attn row cached in LDS
// ---------------------------------------------------------------------------
__global__ void sav_kernel(const float* __restrict__ attn,
                           const float* __restrict__ v,
                           float* __restrict__ sa, int S, int DV) {
    extern __shared__ float as_[];
    int b = blockIdx.y;
    for (int j = threadIdx.x; j < S; j += blockDim.x)
        as_[j] = attn[(size_t)b * S + j];
    __syncthreads();
    int d = blockIdx.x * blockDim.x + threadIdx.x;
    float acc = 0.0f;
    const float* vb = v + (size_t)b * S * DV + d;
    for (int j = 0; j < S; ++j) acc = fmaf(as_[j], vb[(size_t)j * DV], acc);
    sa[(size_t)b * DV + d] = acc;
}

// ---------------------------------------------------------------------------
// K8: out[b,:] = LayerNorm(x1[b,:] + x2[b,:])  (biased var, no eps, no affine)
// ---------------------------------------------------------------------------
__global__ void add_ln_kernel(const float* __restrict__ x1, long long ld1,
                              const float* __restrict__ x2, long long ld2,
                              float* __restrict__ out, int D) {
    int b = blockIdx.x, tid = threadIdx.x;
    __shared__ float red[256];
    __shared__ float stat;
    float v[4], s = 0.0f;
#pragma unroll
    for (int i = 0; i < 4; ++i) {
        int d = tid + i * 256;
        float t = x1[(size_t)b * ld1 + d] + x2[(size_t)b * ld2 + d];
        v[i] = t; s += t;
    }
    red[tid] = s; __syncthreads();
    for (int off = 128; off > 0; off >>= 1) {
        if (tid < off) red[tid] += red[tid + off];
        __syncthreads();
    }
    if (tid == 0) stat = red[0] / (float)D;
    __syncthreads();
    float mu = stat; s = 0.0f;
#pragma unroll
    for (int i = 0; i < 4; ++i) { float t = v[i] - mu; s += t * t; }
    __syncthreads();
    red[tid] = s; __syncthreads();
    for (int off = 128; off > 0; off >>= 1) {
        if (tid < off) red[tid] += red[tid + off];
        __syncthreads();
    }
    if (tid == 0) stat = 1.0f / sqrtf(red[0] / (float)D);
    __syncthreads();
    float inv = stat;
#pragma unroll
    for (int i = 0; i < 4; ++i)
        out[(size_t)b * D + tid + i * 256] = (v[i] - mu) * inv;
}

// ---------------------------------------------------------------------------
// Host launch
// ---------------------------------------------------------------------------
extern "C" void kernel_launch(void* const* d_in, const int* in_sizes, int n_in,
                              void* d_out, int out_size, void* d_ws, size_t ws_size,
                              hipStream_t stream) {
    (void)in_sizes; (void)n_in; (void)out_size; (void)ws_size;

    const int*   x    = (const int*)  d_in[0];
    const float* emb  = (const float*)d_in[1];
    const float* pos  = (const float*)d_in[2];
    const float* wq   = (const float*)d_in[3];
    const float* bq   = (const float*)d_in[4];
    const float* wk   = (const float*)d_in[5];
    const float* bk   = (const float*)d_in[6];
    const float* wv   = (const float*)d_in[7];
    const float* bv   = (const float*)d_in[8];
    const float* w1   = (const float*)d_in[9];
    const float* b1   = (const float*)d_in[10];
    const float* w2   = (const float*)d_in[11];
    const float* b2   = (const float*)d_in[12];
    const float* wout = (const float*)d_in[13];
    const float* bout = (const float*)d_in[14];
    float* out = (float*)d_out;

    // --- workspace carving (256B aligned bump allocator) ---
    char* p = (char*)d_ws;
    auto alloc = [&](size_t bytes) -> char* {
        char* r = p;
        p += (bytes + 255) & ~(size_t)255;
        return r;
    };
    const size_t rows = (size_t)B_ * S_;
    bf16_t* hb   = (bf16_t*)alloc(rows * D_ * sizeof(bf16_t));     // 16 MB
    float*  hf   = (float*) alloc(rows * D_ * sizeof(float));      // 32 MB
    bf16_t* wkb  = (bf16_t*)alloc((size_t)D_ * DK_ * sizeof(bf16_t)); // [N,K]
    bf16_t* wvb  = (bf16_t*)alloc((size_t)D_ * DV_ * sizeof(bf16_t)); // [N,K]
    float*  kf   = (float*) alloc(rows * DK_ * sizeof(float));     // 32 MB
    float*  vf   = (float*) alloc(rows * DV_ * sizeof(float));     // 32 MB
    float*  qbuf = (float*) alloc((size_t)B_ * DK_ * sizeof(float));
    float*  sc   = (float*) alloc((size_t)B_ * S_  * sizeof(float));
    float*  at   = (float*) alloc((size_t)B_ * S_  * sizeof(float));
    float*  sa   = (float*) alloc((size_t)B_ * DV_ * sizeof(float));
    float*  zb   = (float*) alloc((size_t)B_ * D_  * sizeof(float));
    float*  ff1  = (float*) alloc((size_t)B_ * DH_ * sizeof(float));
    float*  ffb  = (float*) alloc((size_t)B_ * D_  * sizeof(float));
    float*  yb   = (float*) alloc((size_t)B_ * D_  * sizeof(float));

    // 1) transpose-convert the two big-GEMM weights to bf16 [N,K]
    {
        dim3 tgrid(DK_ / 32, D_ / 32);
        cvt_transpose_kernel<<<tgrid, 256, 0, stream>>>(wk, wkb, D_, DK_);
        cvt_transpose_kernel<<<tgrid, 256, 0, stream>>>(wv, wvb, D_, DV_);
    }
    // 2) embedding + positional add
    embed_kernel<<<(int)rows, 256, 0, stream>>>(x, emb, pos, hf, hb, S_, D_);

    // 3) k = h@wk + bk ; v = h@wv + bv   (bf16 WMMA + pipelined TDM staging)
    {
        dim3 grid(DK_ / BN, (int)(rows / BM));
        gemm_bf16_wmma<<<grid, 256, 0, stream>>>(hb, wkb, bk, kf,
                                                 (int)rows, DK_, D_);
        gemm_bf16_wmma<<<grid, 256, 0, stream>>>(hb, wvb, bv, vf,
                                                 (int)rows, DV_, D_);
    }
    // 4) q at last token only: q[b,:] = h[b,S-1,:]@wq + bq
    gemv_row_kernel<<<dim3(DK_ / 256, B_), 256, D_ * sizeof(float), stream>>>(
        hf + (size_t)(S_ - 1) * D_, (long long)S_ * D_,
        wq, bq, qbuf, DK_, D_, DK_, 0);
    // 5) scores[b,j] = q.k / sqrt(DK)
    scores_kernel<<<dim3(S_, B_), 256, 0, stream>>>(qbuf, kf, sc, S_, DK_,
                                                    1.0f / 32.0f);
    // 6) softmax
    softmax_kernel<<<B_, 256, 0, stream>>>(sc, at, S_);
    // 7) sa = attn @ v
    sav_kernel<<<dim3(DV_ / 256, B_), 256, S_ * sizeof(float), stream>>>(
        at, vf, sa, S_, DV_);
    // 8) z = LN(h_last + sa)
    add_ln_kernel<<<B_, 256, 0, stream>>>(hf + (size_t)(S_ - 1) * D_,
                                          (long long)S_ * D_,
                                          sa, (long long)DV_, zb, D_);
    // 9) ff1 = z@w1 + b1
    gemv_row_kernel<<<dim3(DH_ / 256, B_), 256, D_ * sizeof(float), stream>>>(
        zb, (long long)D_, w1, b1, ff1, DH_, D_, DH_, 0);
    // 10) ff = gelu(ff1@w2 + b2)
    gemv_row_kernel<<<dim3(D_ / 256, B_), 256, DH_ * sizeof(float), stream>>>(
        ff1, (long long)DH_, w2, b2, ffb, D_, DH_, D_, 1);
    // 11) y = LN(z + ff)
    add_ln_kernel<<<B_, 256, 0, stream>>>(zb, (long long)D_,
                                          ffb, (long long)D_, yb, D_);
    // 12) out = y @ wout + bout   (streams 205 MB of wout: the HBM bound)
    gemv_row_kernel<<<dim3((V_ + 255) / 256, B_), 256, D_ * sizeof(float),
                     stream>>>(yb, (long long)D_, wout, bout, out, V_, D_, V_, 0);
}
